// PointUNetObj_53695681134619
// MI455X (gfx1250) — compile-verified, matrix operand order checked by reading, not run on hardware
//
#include <hip/hip_runtime.h>
#include <hip/hip_bf16.h>

typedef __attribute__((ext_vector_type(16))) _Float16 v16h;
typedef __attribute__((ext_vector_type(8)))  _Float16 v8h;
typedef __attribute__((ext_vector_type(8)))  float    v8f;

#define NTHREADS 128   // 4 waves (wave32)
#define NWAVES   4
#define SA_BUF   9216  // halves per LDS ping-pong buffer (18KB each)
#define MAXROWS  64

// ---------------------------------------------------------------------------
// WMMA fragment loader: row-major matrix with stride (halves), 16x32 f16 frag.
// Lane l holds row (l&15); half-wave h covers K = k0 + {h*8..h*8+7, 16+h*8..}.
// Matches CDNA5 ISA 16-bit A/B layout (CDNA5 ISA 7.12.2).
// ---------------------------------------------------------------------------
__device__ __forceinline__ v16h load_frag16(const _Float16* base, int rowIdx,
                                            int strideH, int k0, int hlf) {
  const _Float16* p = base + (size_t)rowIdx * strideH + k0 + hlf * 8;
  v8h lo = *reinterpret_cast<const v8h*>(p);
  v8h hi = *reinterpret_cast<const v8h*>(p + 16);
  v16h a;
#pragma unroll
  for (int i = 0; i < 8; ++i) { a[i] = lo[i]; a[i + 8] = hi[i]; }
  return a;
}

#define WMMA_F16(AF, BF, ACC) \
  __builtin_amdgcn_wmma_f32_16x16x32_f16(false, (AF), false, (BF), (short)0, (ACC), false, false)

// ---------------------------------------------------------------------------
// One MLP layer over R rows (R mult of 16) held in LDS (f16, stride sIn, sIn
// mult of 32, zero padded). Weights W: f16 [cout x sIn] zero padded.
// 2-wide N-tile blocking (A fragment reused for 2 WMMAs). The K loop is an
// explicit 2-deep register double-buffer (sets A/B, unrolled x2 with mid-loop
// exits) so there are NO rotation copies and NO zero-init of spare buffers:
// per 32-K slab the loop body is just 2 ds_load_b128 + 4 global_load_b128 +
// 2 prefetches + 2 WMMAs.
// If Obuf != null: epilogue bias+BN+ReLU -> f16 LDS out (stride cout).
// Else: final layer -> ReLU + max over rows -> atomicMax into gOut[gBase+col].
// ---------------------------------------------------------------------------
__device__ __forceinline__ void sa_gemm(const _Float16* A, int sIn, int R,
                                        const _Float16* W, const float* bias,
                                        const float* gain, const float* beta,
                                        int cout, _Float16* Obuf, float* gOut,
                                        int gBase, int lane, int hlf, int wave) {
  const int nRowT = R >> 4;
  const int nPair = (cout >> 4) >> 1;  // col tiles processed in pairs (always even)
  const int nT = nRowT * nPair;
  for (int t = wave; t < nT; t += NWAVES) {
    const int tm = t % nRowT;
    const int tp = t / nRowT;
    const int arow  = tm * 16 + (lane & 15);
    const int bcol0 = tp * 32 + (lane & 15);
    const int bcol1 = bcol0 + 16;
    v8f acc0 = {}, acc1 = {};
    // stage A: slab 0
    v16h aA  = load_frag16(A, arow,  sIn, 0, hlf);
    v16h bA0 = load_frag16(W, bcol0, sIn, 0, hlf);
    v16h bA1 = load_frag16(W, bcol1, sIn, 0, hlf);
    v16h aB = aA, bB0 = bA0, bB1 = bA1;  // one-time init (silences uninit paths)
    int k0 = 0;
    while (true) {
      const int kn = k0 + 32;
      if (kn < sIn) {  // issue stage B loads for next slab before MMA on A
        aB  = load_frag16(A, arow,  sIn, kn, hlf);
        bB0 = load_frag16(W, bcol0, sIn, kn, hlf);
        bB1 = load_frag16(W, bcol1, sIn, kn, hlf);
        __builtin_prefetch(W + (size_t)bcol0 * sIn + kn + 32, 0, 3);
        __builtin_prefetch(W + (size_t)bcol1 * sIn + kn + 32, 0, 3);
      }
      acc0 = WMMA_F16(aA, bA0, acc0);
      acc1 = WMMA_F16(aA, bA1, acc1);
      if (kn >= sIn) break;
      const int k2 = kn + 32;
      if (k2 < sIn) {  // issue stage A loads before MMA on B
        aA  = load_frag16(A, arow,  sIn, k2, hlf);
        bA0 = load_frag16(W, bcol0, sIn, k2, hlf);
        bA1 = load_frag16(W, bcol1, sIn, k2, hlf);
        __builtin_prefetch(W + (size_t)bcol0 * sIn + k2 + 32, 0, 3);
        __builtin_prefetch(W + (size_t)bcol1 * sIn + k2 + 32, 0, 3);
      }
      acc0 = WMMA_F16(aB, bB0, acc0);
      acc1 = WMMA_F16(aB, bB1, acc1);
      if (k2 >= sIn) break;
      k0 = k2;
    }
    const int col0 = tp * 32 + (lane & 15);
    const int col1 = col0 + 16;
    const float rs = rsqrtf(1.0f + 1e-5f);
    const float sc0 = gain[col0] * rs, bi0 = bias[col0], be0 = beta[col0];
    const float sc1 = gain[col1] * rs, bi1 = bias[col1], be1 = beta[col1];
    if (Obuf) {
#pragma unroll
      for (int r = 0; r < 8; ++r) {
        const int row = tm * 16 + r + hlf * 8;  // C/D layout: M = r (+8 hi half)
        float y0 = fmaxf((acc0[r] + bi0) * sc0 + be0, 0.0f);
        float y1 = fmaxf((acc1[r] + bi1) * sc1 + be1, 0.0f);
        Obuf[(size_t)row * cout + col0] = (_Float16)y0;
        Obuf[(size_t)row * cout + col1] = (_Float16)y1;
      }
    } else {
      float m0 = 0.0f, m1 = 0.0f;  // ReLU lower bound
#pragma unroll
      for (int r = 0; r < 8; ++r) {
        m0 = fmaxf(m0, fmaxf((acc0[r] + bi0) * sc0 + be0, 0.0f));
        m1 = fmaxf(m1, fmaxf((acc1[r] + bi1) * sc1 + be1, 0.0f));
      }
      m0 = fmaxf(m0, __shfl_xor(m0, 16, 32));  // merge half-waves (same N)
      m1 = fmaxf(m1, __shfl_xor(m1, 16, 32));
      if (lane < 16) {
        atomicMax(reinterpret_cast<int*>(gOut) + gBase + col0, __float_as_int(m0));
        atomicMax(reinterpret_cast<int*>(gOut) + gBase + col1, __float_as_int(m1));
      }
    }
  }
}

// ---------------------------------------------------------------------------
// Fused set-abstraction MLP: gather grouped points (ball-query or group_all)
// into LDS as f16, run 3 MLP layers with WMMA, fuse max-pool via atomicMax.
// grid = B * S * (K / rows); block = 128 threads.
// ---------------------------------------------------------------------------
struct SAParams {
  const float* xyz;      // [B,N,3]
  const float* pts;      // [B,N,Cpts]
  const float* newxyz;   // [B,S,3] or null (group_all)
  const int*   idx;      // [B,S,K] or null (group_all -> id=k)
  const _Float16 *w0, *w1, *w2;
  const float *b0, *g0, *e0, *b1, *g1, *e1, *b2, *g2, *e2;
  float* out;            // [B,S,cout2], pre-zeroed (atomic max target)
  int B, N, S, K, Cin, Cpts, rows;
  int cout0, cout1, cout2;
};

__global__ void __launch_bounds__(NTHREADS) sa_fused_wmma(SAParams p) {
  __shared__ _Float16 sBuf0[SA_BUF];
  __shared__ _Float16 sBuf1[SA_BUF];
  __shared__ int sIdx[MAXROWS];
  const int tid = threadIdx.x;
  const int lane = tid & 31;
  const int hlf = (lane >> 4) & 1;
  const int wave = tid >> 5;
  const int bpg = p.K / p.rows;
  const int g = blockIdx.x / bpg;
  const int rb = blockIdx.x % bpg;
  const int b = g / p.S, s = g % p.S;
  const int row0 = rb * p.rows;
  const int s0 = (p.Cin + 31) & ~31;

  // Stage the group's point ids once.
  if (tid < p.rows)
    sIdx[tid] = p.idx ? p.idx[(size_t)(b * p.S + s) * p.K + row0 + tid] : (row0 + tid);
  __syncthreads();

  float cx = 0.f, cy = 0.f, cz = 0.f;
  if (p.newxyz) {
    const float* c = p.newxyz + (size_t)(b * p.S + s) * 3;
    cx = c[0]; cy = c[1]; cz = c[2];
  }
  // Stage grouped input tile (rows x s0, zero padded) into LDS as f16.
  const int total = p.rows * s0;
  for (int i = tid; i < total; i += NTHREADS) {
    const int kl = i / s0;
    const int c = i - kl * s0;
    const int id = sIdx[kl];
    float v = 0.0f;
    if (c < 3) {
      v = p.xyz[((size_t)b * p.N + id) * 3 + c] - (c == 0 ? cx : (c == 1 ? cy : cz));
    } else if (c < p.Cin) {
      v = p.pts[((size_t)b * p.N + id) * p.Cpts + (c - 3)];
    }
    sBuf0[i] = (_Float16)v;
  }
  __syncthreads();

  sa_gemm(sBuf0, s0, p.rows, p.w0, p.b0, p.g0, p.e0, p.cout0, sBuf1, nullptr, 0, lane, hlf, wave);
  __syncthreads();
  sa_gemm(sBuf1, p.cout0, p.rows, p.w1, p.b1, p.g1, p.e1, p.cout1, sBuf0, nullptr, 0, lane, hlf, wave);
  __syncthreads();
  sa_gemm(sBuf0, p.cout1, p.rows, p.w2, p.b2, p.g2, p.e2, p.cout2, nullptr, p.out,
          (b * p.S + s) * p.cout2, lane, hlf, wave);
}

// ---------------------------------------------------------------------------
// Small FC GEMM via WMMA: Y[M,Cout] = relu?bn?(X[M,Cin] @ W^T + b). M <= 16.
// W prepped as f16 [coutPad x Cin], Cin multiple of 32.
// ---------------------------------------------------------------------------
__global__ void __launch_bounds__(NTHREADS)
fc_wmma(const float* __restrict__ X, const _Float16* __restrict__ W,
        const float* __restrict__ bias, const float* __restrict__ gain,
        const float* __restrict__ beta, float* __restrict__ Y,
        int M, int Cin, int Cout, int coutPad, int bnrelu) {
  const int lane = threadIdx.x & 31;
  const int hlf = (lane >> 4) & 1;
  const int wave = threadIdx.x >> 5;
  const int nColT = coutPad >> 4;
  for (int t = blockIdx.x * NWAVES + wave; t < nColT; t += gridDim.x * NWAVES) {
    v8f acc = {};
    const int arow = lane & 15;
    const int bcol = t * 16 + (lane & 15);
    for (int k0 = 0; k0 < Cin; k0 += 32) {
      v16h a;
      if (arow < M) {
        const float* pp = X + (size_t)arow * Cin + k0 + hlf * 8;
#pragma unroll
        for (int i = 0; i < 8; ++i) { a[i] = (_Float16)pp[i]; a[i + 8] = (_Float16)pp[i + 16]; }
      } else {
#pragma unroll
        for (int i = 0; i < 16; ++i) a[i] = (_Float16)0.f;
      }
      v16h bf = load_frag16(W, bcol, Cin, k0, hlf);
      __builtin_prefetch(W + (size_t)bcol * Cin + k0 + 32, 0, 3);
      acc = WMMA_F16(a, bf, acc);
    }
    const int col = t * 16 + (lane & 15);
#pragma unroll
    for (int r = 0; r < 8; ++r) {
      const int row = r + hlf * 8;
      if (row < M && col < Cout) {
        float y = acc[r];
        if (bnrelu) {
          y = (y + bias[col]) * (gain[col] * rsqrtf(1.0f + 1e-5f)) + beta[col];
          y = fmaxf(y, 0.0f);
        } else {
          y += bias[col];
        }
        Y[(size_t)row * Cout + col] = y;
      }
    }
  }
}

// --------------------------- helper kernels --------------------------------
__global__ void prep_w_kernel(const float* __restrict__ w, _Float16* __restrict__ o,
                              int cout, int cin, int coutPad, int sIn) {
  const int i = blockIdx.x * blockDim.x + threadIdx.x;
  const int total = coutPad * sIn;
  if (i >= total) return;
  const int r = i / sIn, c = i - r * sIn;
  o[i] = (r < cout && c < cin) ? (_Float16)w[(size_t)r * cin + c] : (_Float16)0.f;
}

__global__ void fill_kernel(float* p, int n) {
  const int i = blockIdx.x * blockDim.x + threadIdx.x;
  if (i < n) p[i] = 0.0f;
}

__global__ void split_xyz_kernel(const float* __restrict__ in, float* __restrict__ x3,
                                 float* __restrict__ f3, int total) {  // total = B*N*3
  const int i = blockIdx.x * blockDim.x + threadIdx.x;
  if (i >= total) return;
  const int pt = i / 3, c = i - pt * 3;
  x3[i] = in[(size_t)pt * 6 + c];
  f3[i] = in[(size_t)pt * 6 + 3 + c];
}

// Iterative farthest point sampling; one block per batch; dist kept in LDS.
__global__ void fps_kernel(const float* __restrict__ xyz, int N, int npoint,
                           int* __restrict__ outIdx) {
  __shared__ float sd[2304];
  __shared__ float sv[256];
  __shared__ int   si[256];
  const int b = blockIdx.x, tid = threadIdx.x;
  const float* base = xyz + (size_t)b * N * 3;
  for (int i = tid; i < N; i += 256) sd[i] = 1e10f;
  __syncthreads();
  int far = 0;
  for (int j = 0; j < npoint; ++j) {
    if (tid == 0) outIdx[b * npoint + j] = far;
    const float cx = base[far * 3], cy = base[far * 3 + 1], cz = base[far * 3 + 2];
    float bv = -1.0f; int bi = 0;
    for (int i = tid; i < N; i += 256) {
      const float dx = base[i * 3] - cx, dy = base[i * 3 + 1] - cy, dz = base[i * 3 + 2] - cz;
      const float nd = fminf(sd[i], dx * dx + dy * dy + dz * dz);
      sd[i] = nd;
      if (nd > bv) { bv = nd; bi = i; }
    }
    sv[tid] = bv; si[tid] = bi;
    __syncthreads();
    for (int off = 128; off > 0; off >>= 1) {
      if (tid < off) {
        if (sv[tid + off] > sv[tid] ||
            (sv[tid + off] == sv[tid] && si[tid + off] < si[tid])) {
          sv[tid] = sv[tid + off]; si[tid] = si[tid + off];
        }
      }
      __syncthreads();
    }
    far = si[0];
    __syncthreads();
  }
}

__global__ void gather_kernel(const float* __restrict__ src, const int* __restrict__ idx,
                              float* __restrict__ dst, int B, int N, int S, int C) {
  const int i = blockIdx.x * blockDim.x + threadIdx.x;
  if (i >= B * S * C) return;
  const int b = i / (S * C);
  const int r = i - b * S * C;
  const int s = r / C, c = r - (r / C) * C;
  dst[i] = src[((size_t)b * N + idx[b * S + s]) * C + c];
}

// First-nsample-in-index-order within radius; pad with first hit (matches
// sort-then-truncate-then-replace-N semantics of the reference).
__global__ void ball_query_kernel(const float* __restrict__ xyz,
                                  const float* __restrict__ nxyz, int* __restrict__ idx,
                                  int B, int N, int S, int nsample, float r2) {
  const int i = blockIdx.x * blockDim.x + threadIdx.x;
  if (i >= B * S) return;
  const int b = i / S, s = i - b * S;
  const float* c = nxyz + (size_t)(b * S + s) * 3;
  const float cx = c[0], cy = c[1], cz = c[2];
  int* o = idx + (size_t)(b * S + s) * nsample;
  int cnt = 0, first = -1;
  for (int n = 0; n < N && cnt < nsample; ++n) {
    const float* p = xyz + ((size_t)b * N + n) * 3;
    const float dx = p[0] - cx, dy = p[1] - cy, dz = p[2] - cz;
    if (dx * dx + dy * dy + dz * dz <= r2) {
      if (first < 0) first = n;
      o[cnt++] = n;
    }
  }
  if (first < 0) first = 0;
  for (; cnt < nsample; ++cnt) o[cnt] = first;
}

__global__ void copy_rows_kernel(const float* __restrict__ src, float* __restrict__ dst,
                                 int B, int Ns, int C, int Nd, int off) {
  const int i = blockIdx.x * blockDim.x + threadIdx.x;
  if (i >= B * Ns * C) return;
  const int b = i / (Ns * C);
  const int r = i - b * Ns * C;
  const int n = r / C, c = r - n * C;
  dst[((size_t)b * Nd + off + n) * C + c] = src[i];
}

__device__ __forceinline__ unsigned hash_u32(unsigned x) {
  x ^= x >> 16; x *= 0x7feb352dU; x ^= x >> 15; x *= 0x846ca68bU; x ^= x >> 16;
  return x;
}

__global__ void rw_xyz_kernel(const float* __restrict__ src, float* __restrict__ dst,
                              int B, int Nsrc, int steps, float sigma) {
  const int i = blockIdx.x * blockDim.x + threadIdx.x;
  if (i >= B * Nsrc * 3) return;
  const float base = src[i];
  const int b = i / (Nsrc * 3);
  const int r = i - b * Nsrc * 3;
  const int n = r / 3, c = r - n * 3;
  float accv = 0.0f;
  for (int s = 0; s < steps; ++s) {
    const unsigned h1 = hash_u32(0x9e3779b9u ^ (unsigned)(i * steps + s));
    const unsigned h2 = hash_u32(h1 + 0x85ebca6bu);
    const float u1 = (h1 >> 8) * (1.0f / 16777216.0f) + 1e-7f;
    const float u2 = (h2 >> 8) * (1.0f / 16777216.0f);
    accv += sigma * sqrtf(-2.0f * logf(u1)) * cosf(6.2831853f * u2);
    dst[((size_t)b * (Nsrc * steps) + (n * steps + s)) * 3 + c] = base + accv;
  }
}

__global__ void repeat_kernel(const float* __restrict__ src, float* __restrict__ dst,
                              int B, int Nsrc, int steps, int C) {
  const long long i = (long long)blockIdx.x * blockDim.x + threadIdx.x;
  const long long total = (long long)B * Nsrc * steps * C;
  if (i >= total) return;
  const int c = (int)(i % C);
  long long r = i / C;
  r /= steps;                      // drop step
  const int n = (int)(r % Nsrc);
  const int b = (int)(r / Nsrc);
  dst[i] = src[((size_t)b * Nsrc + n) * C + c];
}

// ---------------------------------------------------------------------------
extern "C" void kernel_launch(void* const* d_in, const int* in_sizes, int n_in,
                              void* d_out, int out_size, void* d_ws, size_t ws_size,
                              hipStream_t stream) {
  (void)in_sizes; (void)n_in; (void)out_size; (void)ws_size;
  int ii = 0;
  auto nf = [&]() { return (const float*)d_in[ii++]; };
  const float* xyz_in = nf();

  struct LParam { const float *w, *b, *g, *be; int cout, cin; };
  LParam sa[5][3];
  const int saCin[5] = {6, 131, 131, 259, 515};
  const int saOut[5][3] = {{64, 64, 128}, {128, 128, 256}, {128, 128, 256},
                           {256, 256, 512}, {512, 512, 1024}};
  for (int i = 0; i < 5; ++i) {
    int c = saCin[i];
    for (int l = 0; l < 3; ++l) {
      sa[i][l].w = nf(); sa[i][l].b = nf(); sa[i][l].g = nf(); sa[i][l].be = nf();
      sa[i][l].cin = c; sa[i][l].cout = saOut[i][l];
      c = saOut[i][l];
    }
  }
  const float* fc1_w = nf(); const float* fc1_b = nf();
  const float* bn1_g = nf(); const float* bn1_b = nf();
  const float* fc2_w = nf(); const float* fc2_b = nf();
  const float* bn2_g = nf(); const float* bn2_b = nf();
  const float* fc3_w = nf(); const float* fc3_b = nf();

  const int B = 4, N0 = 2048;
  auto c32 = [](int x) { return (x + 31) & ~31; };
  size_t off = 0;
  char* wsb = (char*)d_ws;
  auto alloc = [&](size_t bytes) -> void* {
    void* r = wsb + off;
    off = (off + bytes + 255) & ~(size_t)255;
    return r;
  };

  // f16 zero-padded weights
  _Float16* w16[5][3];
  for (int i = 0; i < 5; ++i)
    for (int l = 0; l < 3; ++l)
      w16[i][l] = (_Float16*)alloc((size_t)sa[i][l].cout * c32(sa[i][l].cin) * 2);
  _Float16* fc1w16 = (_Float16*)alloc((size_t)512 * 1024 * 2);
  _Float16* fc2w16 = (_Float16*)alloc((size_t)256 * 512 * 2);
  _Float16* fc3w16 = (_Float16*)alloc((size_t)48 * 256 * 2);

  auto prep = [&](const float* w, _Float16* o, int cout, int cin, int coutPad) {
    const int sIn = c32(cin);
    const int tot = coutPad * sIn;
    prep_w_kernel<<<(tot + 255) / 256, 256, 0, stream>>>(w, o, cout, cin, coutPad, sIn);
  };
  for (int i = 0; i < 5; ++i)
    for (int l = 0; l < 3; ++l)
      prep(sa[i][l].w, w16[i][l], sa[i][l].cout, sa[i][l].cin, sa[i][l].cout);
  prep(fc1_w, fc1w16, 512, 1024, 512);
  prep(fc2_w, fc2w16, 256, 512, 256);
  prep(fc3_w, fc3w16, 42, 256, 48);

  // activation / index buffers
  float* l0x = (float*)alloc((size_t)B * N0 * 3 * 4);
  float* l0p = (float*)alloc((size_t)B * N0 * 3 * 4);
  int*   idx1 = (int*)alloc((size_t)B * 512 * 4);
  float* l1x = (float*)alloc((size_t)B * 512 * 3 * 4);
  int*   bq1 = (int*)alloc((size_t)B * 512 * 64 * 4);
  float* l1p = (float*)alloc((size_t)B * 512 * 128 * 4);
  int*   idx5 = (int*)alloc((size_t)B * 128 * 4);
  float* l5x = (float*)alloc((size_t)B * 128 * 3 * 4);
  int*   bq5 = (int*)alloc((size_t)B * 128 * 64 * 4);
  float* l5p = (float*)alloc((size_t)B * 128 * 256 * 4);
  int*   pidx = (int*)alloc((size_t)B * 256 * 4);
  float* l2inx = (float*)alloc((size_t)B * 256 * 3 * 4);
  float* l2inp = (float*)alloc((size_t)B * 256 * 128 * 4);
  int*   idx2 = (int*)alloc((size_t)B * 32 * 4);
  float* l2x = (float*)alloc((size_t)B * 32 * 3 * 4);
  int*   bq2 = (int*)alloc((size_t)B * 32 * 64 * 4);
  float* l2p = (float*)alloc((size_t)B * 32 * 256 * 4);
  float* upx = (float*)alloc((size_t)B * 2048 * 3 * 4);
  float* upp = (float*)alloc((size_t)B * 2048 * 256 * 4);
  float* l3inx = (float*)alloc((size_t)B * 2176 * 3 * 4);
  float* l3inp = (float*)alloc((size_t)B * 2176 * 256 * 4);
  int*   idx3 = (int*)alloc((size_t)B * 512 * 4);
  float* l3x = (float*)alloc((size_t)B * 512 * 3 * 4);
  int*   bq3 = (int*)alloc((size_t)B * 512 * 64 * 4);
  float* l3p = (float*)alloc((size_t)B * 512 * 512 * 4);
  float* l4 = (float*)alloc((size_t)B * 1024 * 4);
  float* h1 = (float*)alloc((size_t)B * 512 * 4);
  float* h2 = (float*)alloc((size_t)B * 256 * 4);

  auto run_sa = [&](const float* xyz, const float* pts, const float* nxyz,
                    const int* idx, int saI, int Nn, int S, int K, int Cpts,
                    int rows, float* out) {
    SAParams p{};
    p.xyz = xyz; p.pts = pts; p.newxyz = nxyz; p.idx = idx;
    p.w0 = w16[saI][0]; p.w1 = w16[saI][1]; p.w2 = w16[saI][2];
    p.b0 = sa[saI][0].b; p.g0 = sa[saI][0].g; p.e0 = sa[saI][0].be;
    p.b1 = sa[saI][1].b; p.g1 = sa[saI][1].g; p.e1 = sa[saI][1].be;
    p.b2 = sa[saI][2].b; p.g2 = sa[saI][2].g; p.e2 = sa[saI][2].be;
    p.out = out; p.B = B; p.N = Nn; p.S = S; p.K = K;
    p.Cin = 3 + Cpts; p.Cpts = Cpts; p.rows = rows;
    p.cout0 = sa[saI][0].cout; p.cout1 = sa[saI][1].cout; p.cout2 = sa[saI][2].cout;
    const int grid = B * S * (K / rows);
    sa_fused_wmma<<<grid, NTHREADS, 0, stream>>>(p);
  };

  // --- split l0 ---
  split_xyz_kernel<<<(B * N0 * 3 + 255) / 256, 256, 0, stream>>>(xyz_in, l0x, l0p, B * N0 * 3);

  // --- sa1 ---
  fps_kernel<<<B, 256, 0, stream>>>(l0x, N0, 512, idx1);
  gather_kernel<<<(B * 512 * 3 + 255) / 256, 256, 0, stream>>>(l0x, idx1, l1x, B, N0, 512, 3);
  ball_query_kernel<<<(B * 512 + 127) / 128, 128, 0, stream>>>(l0x, l1x, bq1, B, N0, 512, 64, 0.04f * 0.04f);
  fill_kernel<<<(B * 512 * 128 + 255) / 256, 256, 0, stream>>>(l1p, B * 512 * 128);
  run_sa(l0x, l0p, l1x, bq1, 0, N0, 512, 64, 3, 64, l1p);

  // --- sa5 ---
  fps_kernel<<<B, 256, 0, stream>>>(l1x, 512, 128, idx5);
  gather_kernel<<<(B * 128 * 3 + 255) / 256, 256, 0, stream>>>(l1x, idx5, l5x, B, 512, 128, 3);
  ball_query_kernel<<<(B * 128 + 127) / 128, 128, 0, stream>>>(l1x, l5x, bq5, B, 512, 128, 64, 0.01f);
  fill_kernel<<<(B * 128 * 256 + 255) / 256, 256, 0, stream>>>(l5p, B * 128 * 256);
  run_sa(l1x, l1p, l5x, bq5, 1, 512, 128, 64, 128, 32, l5p);

  // --- downsample for sa2 ---
  fps_kernel<<<B, 256, 0, stream>>>(l1x, 512, 256, pidx);
  gather_kernel<<<(B * 256 * 3 + 255) / 256, 256, 0, stream>>>(l1x, pidx, l2inx, B, 512, 256, 3);
  gather_kernel<<<(B * 256 * 128 + 255) / 256, 256, 0, stream>>>(l1p, pidx, l2inp, B, 512, 256, 128);

  // --- sa2 ---
  fps_kernel<<<B, 256, 0, stream>>>(l2inx, 256, 32, idx2);
  gather_kernel<<<(B * 32 * 3 + 255) / 256, 256, 0, stream>>>(l2inx, idx2, l2x, B, 256, 32, 3);
  ball_query_kernel<<<(B * 32 + 127) / 128, 128, 0, stream>>>(l2inx, l2x, bq2, B, 256, 32, 64, 0.01f);
  fill_kernel<<<(B * 32 * 256 + 255) / 256, 256, 0, stream>>>(l2p, B * 32 * 256);
  run_sa(l2inx, l2inp, l2x, bq2, 2, 256, 32, 64, 128, 32, l2p);

  // --- random walk upsample ---
  rw_xyz_kernel<<<(B * 32 * 3 + 255) / 256, 256, 0, stream>>>(l2x, upx, B, 32, 64, 0.05f);
  {
    const long long tot = (long long)B * 32 * 64 * 256;
    repeat_kernel<<<(unsigned)((tot + 255) / 256), 256, 0, stream>>>(l2p, upp, B, 32, 64, 256);
  }

  // --- concat l5 + upsampled ---
  copy_rows_kernel<<<(B * 128 * 3 + 255) / 256, 256, 0, stream>>>(l5x, l3inx, B, 128, 3, 2176, 0);
  copy_rows_kernel<<<(B * 2048 * 3 + 255) / 256, 256, 0, stream>>>(upx, l3inx, B, 2048, 3, 2176, 128);
  copy_rows_kernel<<<(B * 128 * 256 + 255) / 256, 256, 0, stream>>>(l5p, l3inp, B, 128, 256, 2176, 0);
  copy_rows_kernel<<<(B * 2048 * 256 + 255) / 256, 256, 0, stream>>>(upp, l3inp, B, 2048, 256, 2176, 128);

  // --- sa3 ---
  fps_kernel<<<B, 256, 0, stream>>>(l3inx, 2176, 512, idx3);
  gather_kernel<<<(B * 512 * 3 + 255) / 256, 256, 0, stream>>>(l3inx, idx3, l3x, B, 2176, 512, 3);
  ball_query_kernel<<<(B * 512 + 127) / 128, 128, 0, stream>>>(l3inx, l3x, bq3, B, 2176, 512, 64, 0.04f);
  fill_kernel<<<(B * 512 * 512 + 255) / 256, 256, 0, stream>>>(l3p, B * 512 * 512);
  run_sa(l3inx, l3inp, l3x, bq3, 3, 2176, 512, 64, 256, 32, l3p);

  // --- sa4 (group_all) ---
  fill_kernel<<<(B * 1024 + 255) / 256, 256, 0, stream>>>(l4, B * 1024);
  run_sa(l3x, l3p, nullptr, nullptr, 4, 512, 1, 512, 512, 16, l4);

  // --- FC head ---
  fc_wmma<<<8, NTHREADS, 0, stream>>>(l4, fc1w16, fc1_b, bn1_g, bn1_b, h1, 4, 1024, 512, 512, 1);
  fc_wmma<<<4, NTHREADS, 0, stream>>>(h1, fc2w16, fc2_b, bn2_g, bn2_b, h2, 4, 512, 256, 256, 1);
  fc_wmma<<<1, NTHREADS, 0, stream>>>(h2, fc3w16, fc3_b, nullptr, nullptr, (float*)d_out, 4, 256, 42, 48, 0);
}
